// CosLoss_22402549416286
// MI455X (gfx1250) — compile-verified, matrix operand order checked by reading, not run on hardware
//
#include <hip/hip_runtime.h>
#include <hip/hip_bf16.h>

typedef __attribute__((ext_vector_type(16))) __bf16 v16bf;
typedef __attribute__((ext_vector_type(8)))  float  v8f;

#define S_SCALE 30.0f
#define MARGIN  0.4f
#define BATCH   8192
#define DIMK    256
#define NCLS    32000
#define BM      128      // rows per block (8 waves x 16 rows)
#define BN      64       // cols per iteration (4 n-tiles)
#define NSPLIT  10
#define COLS_PER_SPLIT (NCLS / NSPLIT)      // 3200
#define NITER   (COLS_PER_SPLIT / BN)       // 50

// sched_group_barrier masks
#define SG_WMMA    0x008
#define SG_DSREAD  0x100

union Frag {
    uint4        q[2];
    unsigned int u[8];
    v16bf        v;
};

// packed f32->bf16 convert: fptrunc pairs, backend packs to v_cvt_pk_bf16_f32
static __device__ __forceinline__ unsigned int pk_bf16(float a, float b) {
    union { __bf16 h[2]; unsigned int u; } c;
    c.h[0] = (__bf16)a;
    c.h[1] = (__bf16)b;
    return c.u;
}

static __device__ __forceinline__ uint4 pack8(float4 f0, float4 f1) {
    uint4 r;
    r.x = pk_bf16(f0.x, f0.y);
    r.y = pk_bf16(f0.z, f0.w);
    r.z = pk_bf16(f1.x, f1.y);
    r.w = pk_bf16(f1.z, f1.w);
    return r;
}

// wave32 xor-butterfly add via ds_swizzle (imm must be literal)
#define SWZ_ADD(s, imm) \
    (s) += __int_as_float(__builtin_amdgcn_ds_swizzle(__float_as_int(s), (imm)))

static __device__ __forceinline__ float dot4(float4 a, float4 b) {
    return a.x * b.x + a.y * b.y + a.z * b.z + a.w * b.w;
}

// ---------------------------------------------------------------------------
// Kernel 1: per-row inv_norm and exact f32 "correct" logit (wave per row)
// ---------------------------------------------------------------------------
__global__ __launch_bounds__(256) void norm_correct_kernel(
    const float* __restrict__ x, const int* __restrict__ labels,
    const float* __restrict__ W, float* __restrict__ invn,
    float* __restrict__ corr)
{
    const int lane = threadIdx.x & 31;
    const int wid  = threadIdx.x >> 5;
    const int row  = blockIdx.x * 8 + wid;

    const float* xr = x + (size_t)row * DIMK + lane * 8;
    float4 a0 = *(const float4*)(xr);
    float4 a1 = *(const float4*)(xr + 4);
    float  ss = dot4(a0, a0) + dot4(a1, a1);
    SWZ_ADD(ss, 0x041f); SWZ_ADD(ss, 0x081f); SWZ_ADD(ss, 0x101f);
    SWZ_ADD(ss, 0x201f); SWZ_ADD(ss, 0x401f);
    float inv = rsqrtf(ss);

    const int lbl = labels[row];
    const float* wr = W + (size_t)lbl * DIMK + lane * 8;
    float4 b0 = *(const float4*)(wr);
    float4 b1 = *(const float4*)(wr + 4);
    float  d  = dot4(a0, b0) + dot4(a1, b1);
    SWZ_ADD(d, 0x041f); SWZ_ADD(d, 0x081f); SWZ_ADD(d, 0x101f);
    SWZ_ADD(d, 0x201f); SWZ_ADD(d, 0x401f);

    if (lane == 0) {
        invn[row] = inv;
        corr[row] = d * inv;
    }
}

// ---------------------------------------------------------------------------
// Kernel 2: fused bf16 WMMA GEMM + online exp row-sum.
// grid = (BATCH/BM, NSPLIT), block = 256 (8 waves). Wave w owns rows
// [bx*BM + w*16, +16); A fragments resident in VGPRs for the whole kernel.
// LDS tile double-buffered; DS-read/WMMA pipelined via sched_group_barrier.
// ---------------------------------------------------------------------------
__global__ __launch_bounds__(256) void gemm_rowsum_kernel(
    const float* __restrict__ x, const float* __restrict__ W,
    const float* __restrict__ invn, float* __restrict__ partial)
{
    // B staged in exact WMMA fragment layout: one 16B run per (nt,kc,r,lane)
    __shared__ uint4 ldsB[2][BN / 16 * 8 * 2 * 32];   // 2 x 32 KB

    const int tid  = threadIdx.x;
    const int lane = tid & 31;
    const int wid  = tid >> 5;
    const int half = lane >> 4;     // 0: K-low run, 1: K-high run
    const int t16  = lane & 15;
    const int rbase = blockIdx.x * BM + wid * 16;
    const int by    = blockIdx.y;

    // ---- resident A fragments (16 rows x 256 K, bf16) ----
    Frag a[8];
    const float* xrow = x + (size_t)(rbase + t16) * DIMK + half * 8;
#pragma unroll
    for (int kc = 0; kc < 8; ++kc) {
        float4 f0 = *(const float4*)(xrow + kc * 32);
        float4 f1 = *(const float4*)(xrow + kc * 32 + 4);
        float4 g0 = *(const float4*)(xrow + kc * 32 + 16);
        float4 g1 = *(const float4*)(xrow + kc * 32 + 20);
        a[kc].q[0] = pack8(f0, f1);
        a[kc].q[1] = pack8(g0, g1);
    }

    // per-lane coefficient S * inv_norm for the 8 accumulator rows
    float coef[8];
#pragma unroll
    for (int v = 0; v < 8; ++v)
        coef[v] = S_SCALE * invn[rbase + v + 8 * half];

    float rsum[8];
#pragma unroll
    for (int v = 0; v < 8; ++v) rsum[v] = 0.0f;

    // staging assignment: 4 threads per column, each covers 64 contiguous K
    const int tn = tid >> 2;   // local column 0..63
    const int tj = tid & 3;    // K quarter
    const float* wsrc =
        W + (size_t)(by * COLS_PER_SPLIT + tn) * DIMK + tj * 64;

    int runoff[8];
#pragma unroll
    for (int rr = 0; rr < 8; ++rr) {
        const int k0 = tj * 64 + rr * 8;
        const int kc = k0 >> 5, r = (k0 >> 4) & 1, hh = (k0 >> 3) & 1;
        runoff[rr] =
            (((tn >> 4) * 8 + kc) * 2 + r) * 32 + ((tn & 15) + hh * 16);
    }

    // ---- stage tile 0 into buffer 0 ----
#pragma unroll
    for (int rr = 0; rr < 8; ++rr) {
        float4 f0 = *(const float4*)(wsrc + rr * 8);
        float4 f1 = *(const float4*)(wsrc + rr * 8 + 4);
        ldsB[0][runoff[rr]] = pack8(f0, f1);
    }

    for (int it = 0; it < NITER; ++it) {
        const int cur = it & 1;
        const bool have_next = (it + 1 < NITER);

        // issue next tile's global loads + convert (registers only, pre-sync)
        uint4 h[8];
        if (have_next) {
            const float* s = wsrc + (size_t)(it + 1) * BN * DIMK;
#pragma unroll
            for (int rr = 0; rr < 8; ++rr) {
                float4 f0 = *(const float4*)(s + rr * 8);
                float4 f1 = *(const float4*)(s + rr * 8 + 4);
                h[rr] = pack8(f0, f1);
            }
        }

        __syncthreads();   // tile[cur] stores visible; prior compute done

        const uint4* buf = &ldsB[cur][0];
        v8f acc[4] = {};
#pragma unroll
        for (int kc = 0; kc < 8; ++kc) {
            Frag b[4];
#pragma unroll
            for (int nt = 0; nt < 4; ++nt) {
                b[nt].q[0] = buf[((nt * 8 + kc) * 2 + 0) * 32 + lane];
                b[nt].q[1] = buf[((nt * 8 + kc) * 2 + 1) * 32 + lane];
            }
#pragma unroll
            for (int nt = 0; nt < 4; ++nt)
                acc[nt] = __builtin_amdgcn_wmma_f32_16x16x32_bf16(
                    false, a[kc].v, false, b[nt].v, (short)0, acc[nt],
                    false, false);
        }

        // scheduling template: keep DS reads one k-chunk ahead of the WMMAs
        //   DS8(kc0), { DS8(kc+1), WMMA4(kc) } x7, WMMA4(kc7)
        __builtin_amdgcn_sched_group_barrier(SG_DSREAD, 8, 0);
#pragma unroll
        for (int kc = 0; kc < 7; ++kc) {
            __builtin_amdgcn_sched_group_barrier(SG_DSREAD, 8, 0);
            __builtin_amdgcn_sched_group_barrier(SG_WMMA, 4, 0);
        }
        __builtin_amdgcn_sched_group_barrier(SG_WMMA, 4, 0);

        // fused exp(S * wf) accumulation
#pragma unroll
        for (int nt = 0; nt < 4; ++nt)
#pragma unroll
            for (int v = 0; v < 8; ++v)
                rsum[v] += __expf(acc[nt][v] * coef[v]);

        // write next tile into the other LDS buffer (post-sync => race-free)
        if (have_next) {
#pragma unroll
            for (int rr = 0; rr < 8; ++rr)
                ldsB[cur ^ 1][runoff[rr]] = h[rr];
        }
    }

    // butterfly reduce across the 16 column-lanes (keeps half-bit intact)
#pragma unroll
    for (int v = 0; v < 8; ++v) {
        float s = rsum[v];
        SWZ_ADD(s, 0x041f); SWZ_ADD(s, 0x081f);
        SWZ_ADD(s, 0x101f); SWZ_ADD(s, 0x201f);
        rsum[v] = s;
    }
    // lane t (t<8) of each half writes row rbase + 8*half + t
    float out = 0.0f;
#pragma unroll
    for (int v = 0; v < 8; ++v) out = (t16 == v) ? rsum[v] : out;
    if (t16 < 8)
        partial[(size_t)by * BATCH + rbase + 8 * half + t16] = out;
}

// ---------------------------------------------------------------------------
// Kernel 3: deterministic finalize (-mean of CosFace log-likelihoods)
// ---------------------------------------------------------------------------
__global__ __launch_bounds__(256) void finalize_kernel(
    const float* __restrict__ corr, const float* __restrict__ partial,
    float* __restrict__ out)
{
    __shared__ float red[256];
    float acc = 0.0f;
    for (int i = threadIdx.x; i < BATCH; i += 256) {
        float rs = 0.0f;
#pragma unroll
        for (int ny = 0; ny < NSPLIT; ++ny)
            rs += partial[(size_t)ny * BATCH + i];
        const float c    = corr[i];
        const float num  = S_SCALE * (c - MARGIN);
        const float excl = rs - __expf(S_SCALE * c);
        acc += num - __logf(__expf(num) + excl);
    }
    red[threadIdx.x] = acc;
    __syncthreads();
    for (int s = 128; s > 0; s >>= 1) {
        if (threadIdx.x < s) red[threadIdx.x] += red[threadIdx.x + s];
        __syncthreads();
    }
    if (threadIdx.x == 0) out[0] = -red[0] / (float)BATCH;
}

// ---------------------------------------------------------------------------
extern "C" void kernel_launch(void* const* d_in, const int* in_sizes, int n_in,
                              void* d_out, int out_size, void* d_ws,
                              size_t ws_size, hipStream_t stream)
{
    const float* x      = (const float*)d_in[0];
    const int*   labels = (const int*)d_in[1];
    const float* W      = (const float*)d_in[2];
    float*       out    = (float*)d_out;

    float* ws      = (float*)d_ws;
    float* invn    = ws;                 // 8192
    float* corr    = ws + BATCH;         // 8192
    float* partial = ws + 2 * BATCH;     // NSPLIT * 8192

    norm_correct_kernel<<<BATCH / 8, 256, 0, stream>>>(x, labels, W, invn,
                                                       corr);
    dim3 grid(BATCH / BM, NSPLIT);
    gemm_rowsum_kernel<<<grid, 256, 0, stream>>>(x, W, invn, partial);
    finalize_kernel<<<1, 256, 0, stream>>>(corr, partial, out);
}
